// _RefAttention_8916352106882
// MI455X (gfx1250) — matrix operand order repaired and checked by reading, not compiled, not run on hardware
//
#include <hip/hip_runtime.h>
#include <hip/hip_bf16.h>

// ---------------------------------------------------------------------------
// Problem constants (B=1)
// ---------------------------------------------------------------------------
constexpr int S_   = 4096;
constexpr int HID_ = 2048;
constexpr int NH_  = 16;
constexpr int NKV_ = 4;
constexpr int HD_  = 128;
constexpr int RD_  = 64;
__device__ constexpr float SCALE_ = 0.08838834764831845f;   // 1/sqrt(128)

// ---------------------------------------------------------------------------
// WMMA vector types (gfx1250, wave32)
// ---------------------------------------------------------------------------
typedef __bf16 v16bf __attribute__((ext_vector_type(16)));
typedef float  v8f   __attribute__((ext_vector_type(8)));

union Frag32B { uint4 u4[2]; v16bf v; };

// A-matrix fragment (16x32 bf16): lane L holds row M=L&15,
//   elements 0..7  -> K = (L>>4)*8 + 0..7
//   elements 8..15 -> K = 16 + (L>>4)*8 + 0..7
// caller passes p = &A[row*lda + k0 + (lane>>4)*8]
__device__ inline v16bf load_afrag(const unsigned short* p) {
  Frag32B f;
  f.u4[0] = *reinterpret_cast<const uint4*>(p);
  f.u4[1] = *reinterpret_cast<const uint4*>(p + 16);
  return f.v;
}

// B-matrix fragment (32x16 bf16): lane L holds col N=L&15,
//   elements 0..7  -> K = (L>>4)*8 + 0..7
//   elements 8..15 -> K = 16 + (L>>4)*8 + 0..7
// caller passes p = &Brow[n*ldb + k0 + (lane>>4)*8]
__device__ inline v16bf load_bfrag(const unsigned short* p) {
  Frag32B f;
  f.u4[0] = *reinterpret_cast<const uint4*>(p);
  f.u4[1] = *reinterpret_cast<const uint4*>(p + 16);
  return f.v;
}

// fp32 -> bf16 round-to-nearest-even (bit manipulation; no __bf16 scalar math)
__device__ inline unsigned short f2bf(float f) {
  unsigned int u = __builtin_bit_cast(unsigned int, f);
  u += 0x7FFFu + ((u >> 16) & 1u);
  return (unsigned short)(u >> 16);
}

// ---------------------------------------------------------------------------
// 1) elementwise fp32 -> bf16
// ---------------------------------------------------------------------------
__global__ void cvt_bf16(const float* __restrict__ src,
                         unsigned short* __restrict__ dst, int n) {
  int i = blockIdx.x * 256 + threadIdx.x;
  if (i < n) dst[i] = f2bf(src[i]);
}

// ---------------------------------------------------------------------------
// 2) GEMM  C[M,N] = A[M,K] @ B[N,K]^T  (bf16 in, f32 out)
//    one wave (32 threads) per 32x64 output tile; weights live in L2.
// ---------------------------------------------------------------------------
__global__ void __launch_bounds__(32)
gemm_bf16_nt(const unsigned short* __restrict__ A,
             const unsigned short* __restrict__ B,
             float* __restrict__ C, int M, int N, int K) {
  const int lane = threadIdx.x;
  const int half = lane >> 4, l16 = lane & 15;
  const int mb = blockIdx.y * 32;
  const int nb = blockIdx.x * 64;

  v8f acc[2][4] = {};
  for (int k0 = 0; k0 < K; k0 += 32) {
    v16bf af[2];
#pragma unroll
    for (int mt = 0; mt < 2; ++mt)
      af[mt] = load_afrag(A + (size_t)(mb + mt * 16 + l16) * K + k0 + half * 8);
#pragma unroll
    for (int nt = 0; nt < 4; ++nt) {
      v16bf bfr = load_bfrag(B + (size_t)(nb + nt * 16 + l16) * K + k0 + half * 8);
#pragma unroll
      for (int mt = 0; mt < 2; ++mt)
        acc[mt][nt] = __builtin_amdgcn_wmma_f32_16x16x32_bf16(
            false, af[mt], false, bfr, (short)0, acc[mt][nt], false, false);
    }
  }
  // C/D layout: lane holds (M = r + 8*half, N = l16)
#pragma unroll
  for (int mt = 0; mt < 2; ++mt)
#pragma unroll
    for (int nt = 0; nt < 4; ++nt)
#pragma unroll
      for (int r = 0; r < 8; ++r)
        C[(size_t)(mb + mt * 16 + r + 8 * half) * N + nb + nt * 16 + l16] =
            acc[mt][nt][r];
}

// ---------------------------------------------------------------------------
// 3) RMSNorm (over full row of width Wd) + RoPE (first 64 dims of each head)
//    writes bf16 head-major [nheads][S][HD]
// ---------------------------------------------------------------------------
__global__ void rmsnorm_rope_pack(const float* __restrict__ X,
                                  const float* __restrict__ w,
                                  const float* __restrict__ cosb,
                                  const float* __restrict__ sinb,
                                  unsigned short* __restrict__ out, int Wd) {
  const int s = blockIdx.x;
  const float* x = X + (size_t)s * Wd;

  float ss = 0.f;
  for (int i = threadIdx.x; i < (int)Wd; i += blockDim.x) {
    float v = x[i];
    ss += v * v;
  }
#pragma unroll
  for (int m = 16; m >= 1; m >>= 1) ss += __shfl_xor(ss, m, 32);

  __shared__ float red[8];
  __shared__ float rstd_sh;
  if ((threadIdx.x & 31) == 0) red[threadIdx.x >> 5] = ss;
  __syncthreads();
  if (threadIdx.x == 0) {
    float t = 0.f;
    const int nw = blockDim.x >> 5;
    for (int i = 0; i < nw; ++i) t += red[i];
    rstd_sh = rsqrtf(t / (float)Wd + 1e-6f);
  }
  __syncthreads();
  const float rstd = rstd_sh;

  for (int i = threadIdx.x; i < Wd; i += blockDim.x) {
    const int h = i >> 7, d = i & 127;
    const float y = x[i] * rstd * w[i];
    float val;
    if (d < RD_) {
      const int pd = (d < 32) ? i + 32 : i - 32;
      const float y2 = x[pd] * rstd * w[pd];
      const float c = cosb[s * RD_ + d];
      const float sn = sinb[s * RD_ + d];
      val = y * c + ((d < 32) ? -y2 : y2) * sn;
    } else {
      val = y;
    }
    out[((size_t)(h * S_ + s) << 7) + d] = f2bf(val);
  }
}

// ---------------------------------------------------------------------------
// 4) V pack+transpose: V[S][NKV*HD] f32 -> Vt[NKV][HD][S] bf16
//    (so PV B-fragments are contiguous over the kv index)
// ---------------------------------------------------------------------------
__global__ void v_pack_t(const float* __restrict__ V,
                         unsigned short* __restrict__ Vt) {
  const int idx = blockIdx.x * 256 + threadIdx.x;   // over S*512
  const int s = idx >> 9, c = idx & 511;            // c = kv*128 + d
  Vt[((size_t)c << 12) + s] = f2bf(V[idx]);
}

// ---------------------------------------------------------------------------
// 5) causal flash attention; 1 wave per (head, 16-row q tile)
//    Q: [NH][S][HD] bf16, K: [NKV][S][HD] bf16, Vt: [NKV][HD][S] bf16
//    Out: [S][NH*HD] bf16 (row-major, ready for O-projection GEMM)
// ---------------------------------------------------------------------------
__global__ void __launch_bounds__(32)
flash_attn_wmma(const unsigned short* __restrict__ Qb,
                const unsigned short* __restrict__ Kb,
                const unsigned short* __restrict__ Vt,
                unsigned short* __restrict__ Ob) {
  const int qt = blockIdx.x;           // 0..S/16-1
  const int h  = blockIdx.y;           // 0..NH-1
  const int qbase = qt * 16;
  const int kvh = h >> 2;              // n_rep = 4
  const int lane = threadIdx.x, half = lane >> 4, l16 = lane & 15;

  __shared__ __attribute__((aligned(16))) unsigned short Pl[16 * 32];

  // Q fragments for the 16x128 q-tile stay resident: 4 A-frags over K=128
  v16bf qf[4];
  const unsigned short* qrow =
      Qb + ((size_t)(h * S_ + qbase + l16)) * HD_ + half * 8;
#pragma unroll
  for (int kk = 0; kk < 4; ++kk) qf[kk] = load_afrag(qrow + kk * 32);

  v8f o[8] = {};                       // O tile: 16 x 128 f32
  float mrow[8], lrow[8];
#pragma unroll
  for (int r = 0; r < 8; ++r) { mrow[r] = -3.0e38f; lrow[r] = 0.f; }

  for (int kv0 = 0; kv0 < qbase + 16; kv0 += 32) {
    // ---- scores: two 16x16 tiles, 4 chained WMMAs each (K-dim = 128) ----
    v8f sc[2] = {};
#pragma unroll
    for (int t = 0; t < 2; ++t) {
      const unsigned short* kr =
          Kb + ((size_t)(kvh * S_ + kv0 + t * 16 + l16)) * HD_ + half * 8;
#pragma unroll
      for (int kk = 0; kk < 4; ++kk) {
        v16bf bfr = load_bfrag(kr + kk * 32);
        sc[t] = __builtin_amdgcn_wmma_f32_16x16x32_bf16(
            false, qf[kk], false, bfr, (short)0, sc[t], false, false);
      }
    }

    // ---- causal mask + online softmax (row stats live in the lane-half
    //      that owns row m = r + 8*half; reduce over the 16 lanes of half)
    float alpha[8];
#pragma unroll
    for (int r = 0; r < 8; ++r) {
      const int q = qbase + r + 8 * half;
      float s0 = (kv0 + l16 <= q)      ? sc[0][r] * SCALE_ : -3.0e38f;
      float s1 = (kv0 + 16 + l16 <= q) ? sc[1][r] * SCALE_ : -3.0e38f;
      float mx = fmaxf(s0, s1);
#pragma unroll
      for (int msk = 1; msk < 16; msk <<= 1)
        mx = fmaxf(mx, __shfl_xor(mx, msk, 32));
      const float mnew = fmaxf(mrow[r], mx);
      const float a  = __expf(mrow[r] - mnew);
      const float p0 = __expf(s0 - mnew);
      const float p1 = __expf(s1 - mnew);
      float ps = p0 + p1;
#pragma unroll
      for (int msk = 1; msk < 16; msk <<= 1) ps += __shfl_xor(ps, msk, 32);
      lrow[r] = lrow[r] * a + ps;
      mrow[r] = mnew;
      alpha[r] = a;
      sc[0][r] = p0;
      sc[1][r] = p1;
    }

    // rescale running O by alpha (same row ownership as C layout)
#pragma unroll
    for (int dt = 0; dt < 8; ++dt)
#pragma unroll
      for (int r = 0; r < 8; ++r) o[dt][r] *= alpha[r];

    // ---- re-layout P (C layout) -> A layout through LDS ----
    __syncthreads();
#pragma unroll
    for (int r = 0; r < 8; ++r) {
      const int m = r + 8 * half;
      Pl[m * 32 + l16]      = f2bf(sc[0][r]);
      Pl[m * 32 + 16 + l16] = f2bf(sc[1][r]);
    }
    __syncthreads();
    v16bf pf = load_afrag(&Pl[l16 * 32 + half * 8]);

    // ---- O += P(16x32) @ V(32x128): 8 WMMAs over d-tiles ----
#pragma unroll
    for (int dt = 0; dt < 8; ++dt) {
      const unsigned short* vr =
          Vt + ((size_t)(kvh * HD_ + dt * 16 + l16)) * S_ + kv0 + half * 8;
      v16bf vf = load_bfrag(vr);
      o[dt] = __builtin_amdgcn_wmma_f32_16x16x32_bf16(
          false, pf, false, vf, (short)0, o[dt], false, false);
    }
  }

  // ---- finalize and emit bf16 rows for the O-projection GEMM ----
  float inv[8];
#pragma unroll
  for (int r = 0; r < 8; ++r) inv[r] = 1.0f / lrow[r];
#pragma unroll
  for (int dt = 0; dt < 8; ++dt)
#pragma unroll
    for (int r = 0; r < 8; ++r) {
      const int row = qbase + r + 8 * half;
      const int col = h * HD_ + dt * 16 + l16;
      Ob[(size_t)row * (NH_ * HD_) + col] = f2bf(o[dt][r] * inv[r]);
    }
}

// ---------------------------------------------------------------------------
// host-side orchestration
// ---------------------------------------------------------------------------
extern "C" void kernel_launch(void* const* d_in, const int* in_sizes, int n_in,
                              void* d_out, int out_size, void* d_ws,
                              size_t ws_size, hipStream_t stream) {
  (void)in_sizes; (void)n_in; (void)out_size; (void)ws_size;
  const float* X    = (const float*)d_in[0];   // [S, HID]
  const float* cosb = (const float*)d_in[1];   // [1, S, RD]
  const float* sinb = (const float*)d_in[2];   // [1, S, RD]
  const float* Wq   = (const float*)d_in[3];   // [2048, 2048]
  const float* Wk   = (const float*)d_in[4];   // [512, 2048]
  const float* Wv   = (const float*)d_in[5];   // [512, 2048]
  const float* Wo   = (const float*)d_in[6];   // [2048, 2048]
  const float* qw   = (const float*)d_in[7];   // [2048]
  const float* kw   = (const float*)d_in[8];   // [512]
  float* out = (float*)d_out;                  // [S, HID]

  char* ws = (char*)d_ws;
  size_t off = 0;
  auto alloc = [&](size_t bytes) -> void* {
    void* p = ws + off;
    off += (bytes + 255) & ~(size_t)255;
    return p;
  };
  unsigned short* Xb   = (unsigned short*)alloc((size_t)S_ * HID_ * 2);
  unsigned short* Wqb  = (unsigned short*)alloc((size_t)HID_ * HID_ * 2);
  unsigned short* Wkb  = (unsigned short*)alloc((size_t)NKV_ * HD_ * HID_ * 2);
  unsigned short* Wvb  = (unsigned short*)alloc((size_t)NKV_ * HD_ * HID_ * 2);
  unsigned short* Wob  = (unsigned short*)alloc((size_t)HID_ * HID_ * 2);
  float*          Qf   = (float*)alloc((size_t)S_ * HID_ * 4);
  float*          Kf   = (float*)alloc((size_t)S_ * NKV_ * HD_ * 4);
  float*          Vf   = (float*)alloc((size_t)S_ * NKV_ * HD_ * 4);
  unsigned short* Qbf  = (unsigned short*)alloc((size_t)S_ * HID_ * 2);
  unsigned short* Kbf  = (unsigned short*)alloc((size_t)S_ * NKV_ * HD_ * 2);
  unsigned short* Vt   = (unsigned short*)alloc((size_t)S_ * NKV_ * HD_ * 2);
  unsigned short* Abuf = (unsigned short*)alloc((size_t)S_ * HID_ * 2);

  // 1) bf16 conversions
  {
    int nX = S_ * HID_;        // 8,388,608
    int nQ = HID_ * HID_;      // 4,194,304
    int nK = NKV_ * HD_ * HID_;// 1,048,576
    cvt_bf16<<<(nX + 255) / 256, 256, 0, stream>>>(X,  Xb,  nX);
    cvt_bf16<<<(nQ + 255) / 256, 256, 0, stream>>>(Wq, Wqb, nQ);
    cvt_bf16<<<(nK + 255) / 256, 256, 0, stream>>>(Wk, Wkb, nK);
    cvt_bf16<<<(nK + 255) / 256, 256, 0, stream>>>(Wv, Wvb, nK);
    cvt_bf16<<<(nQ + 255) / 256, 256, 0, stream>>>(Wo, Wob, nQ);
  }

  // 2) QKV projections (A@B^T, f32 accumulate)
  gemm_bf16_nt<<<dim3(HID_ / 64, S_ / 32), 32, 0, stream>>>(
      Xb, Wqb, Qf, S_, HID_, HID_);
  gemm_bf16_nt<<<dim3((NKV_ * HD_) / 64, S_ / 32), 32, 0, stream>>>(
      Xb, Wkb, Kf, S_, NKV_ * HD_, HID_);
  gemm_bf16_nt<<<dim3((NKV_ * HD_) / 64, S_ / 32), 32, 0, stream>>>(
      Xb, Wvb, Vf, S_, NKV_ * HD_, HID_);

  // 3) RMSNorm + RoPE -> head-major bf16
  rmsnorm_rope_pack<<<S_, 256, 0, stream>>>(Qf, qw, cosb, sinb, Qbf, HID_);
  rmsnorm_rope_pack<<<S_, 256, 0, stream>>>(Kf, kw, cosb, sinb, Kbf, NKV_ * HD_);

  // 4) V -> bf16 transposed [NKV][HD][S]
  v_pack_t<<<(S_ * NKV_ * HD_) / 256, 256, 0, stream>>>(Vf, Vt);

  // 5) causal flash attention (WMMA QK^T and PV)
  flash_attn_wmma<<<dim3(S_ / 16, NH_), 32, 0, stream>>>(Qbf, Kbf, Vt, Abuf);

  // 6) output projection -> f32 result
  gemm_bf16_nt<<<dim3(HID_ / 64, S_ / 32), 32, 0, stream>>>(
      Abuf, Wob, out, S_, HID_, HID_);
}
